// DecoderStepWithCache_49529562858439
// MI455X (gfx1250) — matrix operand order repaired and checked by reading, not compiled, not run on hardware
//
#include <hip/hip_runtime.h>

#define LNUM   12
#define HNUM   16
#define DHEAD  64
#define DMODEL 1024
#define DFFN   4096
#define MAXLEN 1024
#define VOCAB  32000
#define T_ENC  1500
#define T_PAD  1504
#define QKSCALE 0.125f
#define KCHUNK 64

typedef __attribute__((ext_vector_type(16))) __bf16 v16bf;
typedef __attribute__((ext_vector_type(8)))  float  v8f;
typedef __attribute__((ext_vector_type(4)))  unsigned int u32x4;
typedef __attribute__((ext_vector_type(8)))  int    i32x8;
typedef __attribute__((ext_vector_type(4)))  int    i32x4;

union Frag16 { v16bf v; u32x4 u[2]; };

__device__ __forceinline__ unsigned short f2bf(float f) {
  unsigned int u = __float_as_uint(f);
  u += 0x7fffu + ((u >> 16) & 1u);       // round-to-nearest-even
  return (unsigned short)(u >> 16);
}

#if __has_builtin(__builtin_amdgcn_tensor_load_to_lds)
#define USE_TDM 1
#else
#define USE_TDM 0
#endif

#if USE_TDM
// Issue one 2-D TDM tile load: tile_h rows x tile_w bf16 elements, packed
// contiguously at LDS byte offset lds_addr; global rows strided by row_stride
// elements. D# packing per CDNA5 ISA §8.3/8.4 (count=1, type=2, data_size=2B).
__device__ __forceinline__ void tdm_load_2d(
    unsigned lds_addr, const void* gptr,
    unsigned tile_w, unsigned tile_h, unsigned row_stride)
{
  unsigned long long ga = (unsigned long long)(size_t)gptr;
  u32x4 g0;
  g0[0] = 1u;                                            // count=1, user mode
  g0[1] = lds_addr;                                      // LDS byte address
  g0[2] = (unsigned)(ga & 0xffffffffu);                  // global_addr[31:0]
  g0[3] = (unsigned)((ga >> 32) & 0x01ffffffu) | (2u << 30); // [56:32] | type=2
  i32x8 g1;
  g1[0] = (int)(1u << 16);                               // data_size=1 (2B)
  g1[1] = (int)((row_stride & 0xffffu) << 16);           // tensor_dim0[15:0]
  g1[2] = (int)(((row_stride >> 16) & 0xffffu) | ((tile_h & 0xffffu) << 16)); // td0 hi | td1 lo
  g1[3] = (int)(((tile_h >> 16) & 0xffffu) | ((tile_w & 0xffffu) << 16));     // td1 hi | tile_dim0
  g1[4] = (int)(tile_h & 0xffffu);                       // tile_dim1 (tile_dim2=0)
  g1[5] = (int)row_stride;                               // tensor_dim0_stride lo
  g1[6] = 0;
  g1[7] = 0;
  i32x4 gz = {0, 0, 0, 0};
#if __clang_major__ >= 23
  i32x8 gz8 = {0, 0, 0, 0, 0, 0, 0, 0};
  __builtin_amdgcn_tensor_load_to_lds(g0, g1, gz, gz, gz8, 0);
#else
  __builtin_amdgcn_tensor_load_to_lds(g0, g1, gz, gz, 0);
#endif
}
#endif

// ---------------- embedding + layernorm ----------------
__global__ __launch_bounds__(256) void embed_ln_kernel(
    const int* __restrict__ tokp, const int* __restrict__ stepp,
    const float* __restrict__ tok_emb, const float* __restrict__ pos_emb,
    const float* __restrict__ g, const float* __restrict__ b,
    float* __restrict__ y)
{
  int tid = threadIdx.x;
  int tok = tokp[0], s = stepp[0];
  __shared__ float red[256];
  float v[4];
  #pragma unroll
  for (int r = 0; r < 4; ++r) {
    int i = tid * 4 + r;
    v[r] = tok_emb[(size_t)tok * DMODEL + i] + pos_emb[(size_t)s * DMODEL + i];
  }
  red[tid] = v[0] + v[1] + v[2] + v[3];
  __syncthreads();
  for (int o = 128; o > 0; o >>= 1) { if (tid < o) red[tid] += red[tid + o]; __syncthreads(); }
  float mean = red[0] * (1.f / DMODEL);
  __syncthreads();
  float d0 = v[0]-mean, d1 = v[1]-mean, d2 = v[2]-mean, d3 = v[3]-mean;
  red[tid] = d0*d0 + d1*d1 + d2*d2 + d3*d3;
  __syncthreads();
  for (int o = 128; o > 0; o >>= 1) { if (tid < o) red[tid] += red[tid + o]; __syncthreads(); }
  float rstd = rsqrtf(red[0] * (1.f / DMODEL) + 1e-5f);
  #pragma unroll
  for (int r = 0; r < 4; ++r) {
    int i = tid * 4 + r;
    y[i] = (v[r] - mean) * rstd * g[i] + b[i];
  }
}

// ---------------- layernorm ----------------
__global__ __launch_bounds__(256) void layernorm_kernel(
    const float* __restrict__ x, const float* __restrict__ g,
    const float* __restrict__ b, float* __restrict__ y)
{
  int tid = threadIdx.x;
  __shared__ float red[256];
  float v[4];
  #pragma unroll
  for (int r = 0; r < 4; ++r) v[r] = x[tid * 4 + r];
  red[tid] = v[0] + v[1] + v[2] + v[3];
  __syncthreads();
  for (int o = 128; o > 0; o >>= 1) { if (tid < o) red[tid] += red[tid + o]; __syncthreads(); }
  float mean = red[0] * (1.f / DMODEL);
  __syncthreads();
  float d0 = v[0]-mean, d1 = v[1]-mean, d2 = v[2]-mean, d3 = v[3]-mean;
  red[tid] = d0*d0 + d1*d1 + d2*d2 + d3*d3;
  __syncthreads();
  for (int o = 128; o > 0; o >>= 1) { if (tid < o) red[tid] += red[tid + o]; __syncthreads(); }
  float rstd = rsqrtf(red[0] * (1.f / DMODEL) + 1e-5f);
  #pragma unroll
  for (int r = 0; r < 4; ++r) {
    int i = tid * 4 + r;
    y[i] = (v[r] - mean) * rstd * g[i] + b[i];
  }
}

// ---------------- generic matvec: y = [res +] act(x @ W + b) ----------------
__global__ __launch_bounds__(128) void matvec_kernel(
    const float* __restrict__ x, const float* __restrict__ W,
    const float* __restrict__ b, const float* __restrict__ res,
    float* __restrict__ y, int Din, int Dout, int relu)
{
  __shared__ float xs[DFFN];
  for (int i = threadIdx.x; i < Din; i += 128) xs[i] = x[i];
  __syncthreads();
  int j = blockIdx.x * 128 + threadIdx.x;
  if (j >= Dout) return;
  __builtin_prefetch(W + j, 0, 1);          // global_prefetch_b8
  float acc = 0.f;
  #pragma unroll 4
  for (int i = 0; i < Din; ++i) acc = fmaf(xs[i], W[(size_t)i * Dout + j], acc);
  acc += b[j];
  if (relu) acc = fmaxf(acc, 0.f);
  if (res)  acc += res[j];
  y[j] = acc;
}

// ---------------- scatter new k/v row into output cache ----------------
__global__ void scatter_kv_kernel(
    const float* __restrict__ k_new, const float* __restrict__ v_new,
    const int* __restrict__ stepp, float* __restrict__ out_k,
    float* __restrict__ out_v, int layer)
{
  int i = blockIdx.x * 256 + threadIdx.x;
  if (i >= DMODEL) return;
  int s = stepp[0];
  int h = i >> 6, d = i & 63;
  size_t off = (((size_t)layer * HNUM + h) * MAXLEN + s) * DHEAD + d;
  out_k[off] = k_new[i];
  out_v[off] = v_new[i];
}

// ---------------- self attention over updated cache (one block per head) ----
__global__ __launch_bounds__(256) void self_attn_kernel(
    const float* __restrict__ q, const float* __restrict__ Kc,
    const float* __restrict__ Vc, const int* __restrict__ stepp,
    float* __restrict__ out)
{
  int h = blockIdx.x, tid = threadIdx.x;
  int s = stepp[0];
  __shared__ float qs[DHEAD];
  __shared__ float sc[MAXLEN];
  __shared__ float red[256];
  if (tid < DHEAD) qs[tid] = q[h * DHEAD + tid];
  __syncthreads();
  const float* Kh = Kc + (size_t)h * MAXLEN * DHEAD;
  const float* Vh = Vc + (size_t)h * MAXLEN * DHEAD;
  float lmax = -1e30f;
  for (int p = tid; p < MAXLEN; p += 256) {
    float acc = -1e30f;
    if (p <= s) {
      const float* kr = Kh + (size_t)p * DHEAD;
      acc = 0.f;
      #pragma unroll 8
      for (int d = 0; d < DHEAD; ++d) acc = fmaf(qs[d], kr[d], acc);
      acc *= QKSCALE;
    }
    sc[p] = acc;
    lmax = fmaxf(lmax, acc);
  }
  red[tid] = lmax; __syncthreads();
  for (int o = 128; o > 0; o >>= 1) { if (tid < o) red[tid] = fmaxf(red[tid], red[tid + o]); __syncthreads(); }
  float m = red[0]; __syncthreads();
  float lsum = 0.f;
  for (int p = tid; p < MAXLEN; p += 256) {
    float e = (p <= s) ? __expf(sc[p] - m) : 0.f;
    sc[p] = e; lsum += e;
  }
  red[tid] = lsum; __syncthreads();
  for (int o = 128; o > 0; o >>= 1) { if (tid < o) red[tid] += red[tid + o]; __syncthreads(); }
  float inv = 1.f / red[0];
  __syncthreads();
  int c = tid >> 6, d = tid & 63;
  int p0 = c * 256;
  int p1 = p0 + 256 < (s + 1) ? p0 + 256 : (s + 1);
  float acc = 0.f;
  for (int p = p0; p < p1; ++p) acc = fmaf(sc[p], Vh[(size_t)p * DHEAD + d], acc);
  red[tid] = acc; __syncthreads();
  if (c == 0) out[h * DHEAD + d] = (red[d] + red[64 + d] + red[128 + d] + red[192 + d]) * inv;
}

// ---------------- cross attention over encoder K/V (one block per head) ----
__global__ __launch_bounds__(256) void cross_attn_kernel(
    const float* __restrict__ q, const float* __restrict__ Ke,
    const float* __restrict__ Ve, float* __restrict__ out)
{
  int h = blockIdx.x, tid = threadIdx.x;
  __shared__ float qs[DHEAD];
  __shared__ float sc[T_PAD];
  __shared__ float red[256];
  if (tid < DHEAD) qs[tid] = q[h * DHEAD + tid];
  __syncthreads();
  float lmax = -1e30f;
  for (int p = tid; p < T_ENC; p += 256) {
    const float* kr = Ke + (size_t)p * DMODEL + h * DHEAD;
    float acc = 0.f;
    #pragma unroll 8
    for (int d = 0; d < DHEAD; ++d) acc = fmaf(qs[d], kr[d], acc);
    acc *= QKSCALE;
    sc[p] = acc;
    lmax = fmaxf(lmax, acc);
  }
  red[tid] = lmax; __syncthreads();
  for (int o = 128; o > 0; o >>= 1) { if (tid < o) red[tid] = fmaxf(red[tid], red[tid + o]); __syncthreads(); }
  float m = red[0]; __syncthreads();
  float lsum = 0.f;
  for (int p = tid; p < T_ENC; p += 256) { float e = __expf(sc[p] - m); sc[p] = e; lsum += e; }
  red[tid] = lsum; __syncthreads();
  for (int o = 128; o > 0; o >>= 1) { if (tid < o) red[tid] += red[tid + o]; __syncthreads(); }
  float inv = 1.f / red[0];
  __syncthreads();
  int c = tid >> 6, d = tid & 63;
  int p0 = c * 376;
  int p1 = p0 + 376 < T_ENC ? p0 + 376 : T_ENC;
  float acc = 0.f;
  for (int p = p0; p < p1; ++p) acc = fmaf(sc[p], Ve[(size_t)p * DMODEL + h * DHEAD + d], acc);
  red[tid] = acc; __syncthreads();
  if (c == 0) out[h * DHEAD + d] = (red[d] + red[64 + d] + red[128 + d] + red[192 + d]) * inv;
}

// ---------------- fp32 -> bf16 convert of padded encoder activations -------
__global__ void cvt_enc_kernel(const float* __restrict__ enc, unsigned short* __restrict__ out)
{
  int idx = blockIdx.x * 256 + threadIdx.x;
  if (idx >= T_PAD * DMODEL) return;
  int row = idx >> 10;
  float v = (row < T_ENC) ? enc[idx] : 0.f;
  out[idx] = f2bf(v);
}

// -------- transpose + bf16 convert of both cross-attn weight matrices ------
__global__ void cvt_w_transpose_kernel(
    const float* __restrict__ Wk, const float* __restrict__ Wv,
    unsigned short* __restrict__ WkT, unsigned short* __restrict__ WvT)
{
  __shared__ float tile[32][33];
  const float* W = blockIdx.z ? Wv : Wk;
  unsigned short* O = blockIdx.z ? WvT : WkT;
  int tx = threadIdx.x, ty = threadIdx.y;
  int c0 = blockIdx.x * 32, r0 = blockIdx.y * 32;
  for (int i = ty; i < 32; i += 8) tile[i][tx] = W[(size_t)(r0 + i) * DMODEL + c0 + tx];
  __syncthreads();
  for (int i = ty; i < 32; i += 8) O[(size_t)(c0 + i) * DMODEL + r0 + tx] = f2bf(tile[tx][i]);
}

// ---------------- bf16 WMMA GEMM with TDM->LDS double-buffered staging -----
// C[M_pad x N] = A[M_pad x K] @ B + bias.  A row-major bf16, BT = B^T
// row-major bf16 ([N x K]).  Block = 8 waves computing a 32x64 macro-tile
// (2x4 grid of 16x16 wave tiles).  K is consumed in KCHUNK slices staged in
// LDS: wave 0 issues TDM tensor loads, waits on tensorcnt, and a workgroup
// barrier publishes the slice; WMMA fragments then come from ds_load_b128.
__global__ __launch_bounds__(256) void gemm_bf16_wmma_kernel(
    const unsigned short* __restrict__ A, const unsigned short* __restrict__ BT,
    const float* __restrict__ bias, float* __restrict__ C, int N, int K)
{
  __shared__ __attribute__((aligned(16))) unsigned short sA[2][32 * KCHUNK];
  __shared__ __attribute__((aligned(16))) unsigned short sB[2][64 * KCHUNK];
  int tid  = threadIdx.x;
  int wave = tid >> 5, lane = tid & 31;
  int wm = wave >> 2, wn = wave & 3;
  int m0 = blockIdx.x * 32;
  int n0 = blockIdx.y * 64;
  int sel = lane >> 4, ln15 = lane & 15;
  int nchunks = K / KCHUNK;
  v8f acc = {};

#if USE_TDM
  bool issuer = (wave == 0);
  if (issuer) {
    tdm_load_2d((unsigned)(size_t)&sA[0][0], A  + (size_t)m0 * K, KCHUNK, 32, K);
    tdm_load_2d((unsigned)(size_t)&sB[0][0], BT + (size_t)n0 * K, KCHUNK, 64, K);
  }
  for (int c = 0; c < nchunks; ++c) {
    int cur = c & 1;
    if (issuer) {
      if (c + 1 < nchunks) {
        int kk = (c + 1) * KCHUNK;
        tdm_load_2d((unsigned)(size_t)&sA[cur ^ 1][0], A  + (size_t)m0 * K + kk, KCHUNK, 32, K);
        tdm_load_2d((unsigned)(size_t)&sB[cur ^ 1][0], BT + (size_t)n0 * K + kk, KCHUNK, 64, K);
#if __has_builtin(__builtin_amdgcn_s_wait_tensorcnt)
        __builtin_amdgcn_s_wait_tensorcnt((short)2);   // chunk c complete
#endif
      } else {
#if __has_builtin(__builtin_amdgcn_s_wait_tensorcnt)
        __builtin_amdgcn_s_wait_tensorcnt((short)0);
#endif
      }
    }
    __syncthreads();
    const unsigned short* a = &sA[cur][(wm * 16 + ln15) * KCHUNK];
    const unsigned short* b = &sB[cur][(wn * 16 + ln15) * KCHUNK];
    #pragma unroll
    for (int kl = 0; kl < KCHUNK; kl += 32) {
      Frag16 af, bfr;
      af.u[0]  = *(const u32x4*)(a + kl + sel * 8);
      af.u[1]  = *(const u32x4*)(a + kl + 16 + sel * 8);
      bfr.u[0] = *(const u32x4*)(b + kl + sel * 8);
      bfr.u[1] = *(const u32x4*)(b + kl + 16 + sel * 8);
      acc = __builtin_amdgcn_wmma_f32_16x16x32_bf16(
          false, af.v, false, bfr.v, (short)0, acc, false, false);
    }
    __syncthreads();
  }
#else
  // Fallback: cooperative global->LDS staging (single-buffered).
  for (int c = 0; c < nchunks; ++c) {
    int kk = c * KCHUNK;
    {
      int e = tid * 8;                           // 8 bf16 per thread for A
      int ar = e >> 6, ac = e & 63;
      *(u32x4*)&sA[0][e] = *(const u32x4*)(A + (size_t)(m0 + ar) * K + kk + ac);
      #pragma unroll
      for (int h = 0; h < 2; ++h) {              // 2x8 bf16 per thread for B
        int eb = (tid + h * 256) * 8;
        int br = eb >> 6, bc = eb & 63;
        *(u32x4*)&sB[0][eb] = *(const u32x4*)(BT + (size_t)(n0 + br) * K + kk + bc);
      }
    }
    __syncthreads();
    const unsigned short* a = &sA[0][(wm * 16 + ln15) * KCHUNK];
    const unsigned short* b = &sB[0][(wn * 16 + ln15) * KCHUNK];
    #pragma unroll
    for (int kl = 0; kl < KCHUNK; kl += 32) {
      Frag16 af, bfr;
      af.u[0]  = *(const u32x4*)(a + kl + sel * 8);
      af.u[1]  = *(const u32x4*)(a + kl + 16 + sel * 8);
      bfr.u[0] = *(const u32x4*)(b + kl + sel * 8);
      bfr.u[1] = *(const u32x4*)(b + kl + 16 + sel * 8);
      acc = __builtin_amdgcn_wmma_f32_16x16x32_bf16(
          false, af.v, false, bfr.v, (short)0, acc, false, false);
    }
    __syncthreads();
  }
#endif

  int n = n0 + wn * 16 + ln15;
  float bv = bias[n];
  int mbase = m0 + wm * 16 + 8 * sel;            // C/D layout: VGPR r -> M = r + 8*sel
  #pragma unroll
  for (int r = 0; r < 8; ++r)
    C[(size_t)(mbase + r) * N + n] = acc[r] + bv;
}

// ---------------- log-softmax over VOCAB ----------------
__global__ __launch_bounds__(256) void log_softmax_kernel(
    const float* __restrict__ in, float* __restrict__ out)
{
  int tid = threadIdx.x;
  __shared__ float red[256];
  float m = -1e30f;
  for (int i = tid; i < VOCAB; i += 256) m = fmaxf(m, in[i]);
  red[tid] = m; __syncthreads();
  for (int o = 128; o > 0; o >>= 1) { if (tid < o) red[tid] = fmaxf(red[tid], red[tid + o]); __syncthreads(); }
  float M = red[0]; __syncthreads();
  float s = 0.f;
  for (int i = tid; i < VOCAB; i += 256) s += __expf(in[i] - M);
  red[tid] = s; __syncthreads();
  for (int o = 128; o > 0; o >>= 1) { if (tid < o) red[tid] += red[tid + o]; __syncthreads(); }
  float lse = M + __logf(red[0]);
  for (int i = tid; i < VOCAB; i += 256) out[i] = in[i] - lse;
}

// ===========================================================================
extern "C" void kernel_launch(void* const* d_in, const int* in_sizes, int n_in,
                              void* d_out, int out_size, void* d_ws, size_t ws_size,
                              hipStream_t stream) {
  (void)in_sizes; (void)n_in; (void)out_size; (void)ws_size;
  const float* enc     = (const float*)d_in[0];
  const int*   tok     = (const int*)d_in[1];
  const float* cache_k = (const float*)d_in[2];
  const float* cache_v = (const float*)d_in[3];
  const int*   step    = (const int*)d_in[4];
  const float* tok_emb = (const float*)d_in[5];
  const float* pos_emb = (const float*)d_in[6];
  const float* emb_g   = (const float*)d_in[7];
  const float* emb_b   = (const float*)d_in[8];
  const float* fln_g   = (const float*)d_in[9];
  const float* fln_b   = (const float*)d_in[10];
  const float* Wcls    = (const float*)d_in[11];
  const float* bcls    = (const float*)d_in[12];
  const float* Wq_sa   = (const float*)d_in[13];
  const float* bq_sa   = (const float*)d_in[14];
  const float* Wk_sa   = (const float*)d_in[15];
  const float* bk_sa   = (const float*)d_in[16];
  const float* Wv_sa   = (const float*)d_in[17];
  const float* bv_sa   = (const float*)d_in[18];
  const float* Wo_sa   = (const float*)d_in[19];
  const float* bo_sa   = (const float*)d_in[20];
  const float* Wq_ca   = (const float*)d_in[21];
  const float* bq_ca   = (const float*)d_in[22];
  const float* Wk_ca   = (const float*)d_in[23];
  const float* bk_ca   = (const float*)d_in[24];
  const float* Wv_ca   = (const float*)d_in[25];
  const float* bv_ca   = (const float*)d_in[26];
  const float* Wo_ca   = (const float*)d_in[27];
  const float* bo_ca   = (const float*)d_in[28];
  const float* W1      = (const float*)d_in[29];
  const float* b1      = (const float*)d_in[30];
  const float* W2      = (const float*)d_in[31];
  const float* b2      = (const float*)d_in[32];
  const float* ln1_g   = (const float*)d_in[33];
  const float* ln1_b   = (const float*)d_in[34];
  const float* ln2_g   = (const float*)d_in[35];
  const float* ln2_b   = (const float*)d_in[36];
  const float* ln3_g   = (const float*)d_in[37];
  const float* ln3_b   = (const float*)d_in[38];

  const size_t CACHE_ELEMS = (size_t)LNUM * HNUM * MAXLEN * DHEAD;
  float* logits = (float*)d_out;
  float* out_k  = logits + VOCAB;
  float* out_v  = out_k + CACHE_ELEMS;

  // workspace carve-out (256B aligned)
  char* wp = (char*)d_ws;
  auto alloc = [&](size_t nbytes) {
    char* p = wp; wp += (nbytes + 255) & ~(size_t)255; return (void*)p;
  };
  float* hidden     = (float*)alloc(DMODEL * 4);
  float* xbuf       = (float*)alloc(DMODEL * 4);
  float* qv         = (float*)alloc(DMODEL * 4);
  float* k_new      = (float*)alloc(DMODEL * 4);
  float* v_new      = (float*)alloc(DMODEL * 4);
  float* attn_out   = (float*)alloc(DMODEL * 4);
  float* mid        = (float*)alloc(DFFN * 4);
  float* logits_raw = (float*)alloc((size_t)VOCAB * 4);
  unsigned short* enc_bf = (unsigned short*)alloc((size_t)T_PAD * DMODEL * 2);
  unsigned short* WkT    = (unsigned short*)alloc((size_t)DMODEL * DMODEL * 2);
  unsigned short* WvT    = (unsigned short*)alloc((size_t)DMODEL * DMODEL * 2);
  float* K_enc = (float*)alloc((size_t)T_PAD * DMODEL * 4);
  float* V_enc = (float*)alloc((size_t)T_PAD * DMODEL * 4);

  // pass the KV cache through to the outputs (scatter inserts the new row)
  hipMemcpyAsync(out_k, cache_k, CACHE_ELEMS * 4, hipMemcpyDeviceToDevice, stream);
  hipMemcpyAsync(out_v, cache_v, CACHE_ELEMS * 4, hipMemcpyDeviceToDevice, stream);

  cvt_enc_kernel<<<(T_PAD * DMODEL + 255) / 256, 256, 0, stream>>>(enc, enc_bf);
  embed_ln_kernel<<<1, 256, 0, stream>>>(tok, step, tok_emb, pos_emb, emb_g, emb_b, hidden);

  const dim3 gemm_grid(T_PAD / 32, DMODEL / 64);   // 47 x 16 blocks, 32x64 tile

  for (int l = 0; l < LNUM; ++l) {
    const size_t wo = (size_t)l * DMODEL * DMODEL;
    // ---- self attention ----
    layernorm_kernel<<<1, 256, 0, stream>>>(hidden, ln1_g + l * DMODEL, ln1_b + l * DMODEL, xbuf);
    matvec_kernel<<<8, 128, 0, stream>>>(xbuf, Wq_sa + wo, bq_sa + l * DMODEL, nullptr, qv, DMODEL, DMODEL, 0);
    matvec_kernel<<<8, 128, 0, stream>>>(xbuf, Wk_sa + wo, bk_sa + l * DMODEL, nullptr, k_new, DMODEL, DMODEL, 0);
    matvec_kernel<<<8, 128, 0, stream>>>(xbuf, Wv_sa + wo, bv_sa + l * DMODEL, nullptr, v_new, DMODEL, DMODEL, 0);
    scatter_kv_kernel<<<4, 256, 0, stream>>>(k_new, v_new, step, out_k, out_v, l);
    self_attn_kernel<<<HNUM, 256, 0, stream>>>(
        qv, out_k + (size_t)l * HNUM * MAXLEN * DHEAD,
        out_v + (size_t)l * HNUM * MAXLEN * DHEAD, step, attn_out);
    matvec_kernel<<<8, 128, 0, stream>>>(attn_out, Wo_sa + wo, bo_sa + l * DMODEL, hidden, hidden, DMODEL, DMODEL, 0);
    // ---- cross attention ----
    layernorm_kernel<<<1, 256, 0, stream>>>(hidden, ln2_g + l * DMODEL, ln2_b + l * DMODEL, xbuf);
    matvec_kernel<<<8, 128, 0, stream>>>(xbuf, Wq_ca + wo, bq_ca + l * DMODEL, nullptr, qv, DMODEL, DMODEL, 0);
    cvt_w_transpose_kernel<<<dim3(32, 32, 2), dim3(32, 8), 0, stream>>>(Wk_ca + wo, Wv_ca + wo, WkT, WvT);
    gemm_bf16_wmma_kernel<<<gemm_grid, 256, 0, stream>>>(enc_bf, WkT, bk_ca + l * DMODEL, K_enc, DMODEL, DMODEL);
    gemm_bf16_wmma_kernel<<<gemm_grid, 256, 0, stream>>>(enc_bf, WvT, bv_ca + l * DMODEL, V_enc, DMODEL, DMODEL);
    cross_attn_kernel<<<HNUM, 256, 0, stream>>>(qv, K_enc, V_enc, attn_out);
    matvec_kernel<<<8, 128, 0, stream>>>(attn_out, Wo_ca + wo, bo_ca + l * DMODEL, hidden, hidden, DMODEL, DMODEL, 0);
    // ---- ffn ----
    layernorm_kernel<<<1, 256, 0, stream>>>(hidden, ln3_g + l * DMODEL, ln3_b + l * DMODEL, xbuf);
    matvec_kernel<<<32, 128, 0, stream>>>(xbuf, W1 + (size_t)l * DMODEL * DFFN, b1 + l * DFFN, nullptr, mid, DMODEL, DFFN, 1);
    matvec_kernel<<<8, 128, 0, stream>>>(mid, W2 + (size_t)l * DFFN * DMODEL, b2 + l * DMODEL, hidden, hidden, DFFN, DMODEL, 0);
  }

  layernorm_kernel<<<1, 256, 0, stream>>>(hidden, fln_g, fln_b, xbuf);
  matvec_kernel<<<250, 128, 0, stream>>>(xbuf, Wcls, bcls, nullptr, logits_raw, DMODEL, VOCAB, 0);
  log_softmax_kernel<<<1, 256, 0, stream>>>(logits_raw, logits);
}
